// VanillaMAPPOSparseCommResidualNormAgent_30597347017042
// MI455X (gfx1250) — matrix operand order repaired and checked by reading, not compiled, not run on hardware
//
#include <hip/hip_runtime.h>
#include <math.h>

// ---------------------------------------------------------------------------
// MAPPO sparse-comm agent, fused for MI455X (gfx1250, wave32, WMMA f16).
// One workgroup (4 wave32) per TWO envs (M=32): every B fragment loaded from
// L2 feeds two v_wmma_f32_16x16x32_f16 (one per env A-tile), doubling
// arithmetic intensity vs L2 to 32 FLOP/byte and halving weight traffic.
// ---------------------------------------------------------------------------

typedef __attribute__((ext_vector_type(16))) _Float16 v16h;
typedef __attribute__((ext_vector_type(8)))  float    v8f;

#define WMMA16(a, b, c) \
  __builtin_amdgcn_wmma_f32_16x16x32_f16(false, (a), false, (b), (short)0, (c), false, false)

// ---- problem constants (from reference) -----------------------------------
static constexpr int BS      = 8192;
static constexpr int NAG     = 16;
static constexpr int NACT    = 32;
static constexpr int EPB     = 2;          // envs per block
static constexpr int ROWS    = NAG * EPB;  // 32 rows staged per block
static constexpr float NEGV  = -1e10f;

// ---- f16 weight workspace layout (element offsets) ------------------------
static constexpr size_t OFF_FC1 = 0;                       // 256x256
static constexpr size_t OFF_WIH = OFF_FC1 + 256 * 256;     // 768x256
static constexpr size_t OFF_WHH = OFF_WIH + 768 * 256;     // 768x256
static constexpr size_t OFF_QKV = OFF_WHH + 768 * 256;     // 80x256 (q32|k32|v8|hg2|pad6)
static constexpr size_t OFF_RG1 = OFF_QKV + 80 * 256;      // 256x288 (K padded 264->288)
static constexpr size_t OFF_PH1 = OFF_RG1 + 256 * 288;     // 256x256
static constexpr size_t OFF_PH2 = OFF_PH1 + 256 * 256;     // 32x256

// ---------------------------------------------------------------------------
// WMMA fragment helpers (layouts per CDNA5 ISA 7.12.2, wave32)
// ---------------------------------------------------------------------------
__device__ __forceinline__ v16h load_a(const _Float16* lds, int k0, int lane) {
  const int r    = lane & 15;
  const int koff = (lane >> 4) << 3;            // 0 or 8
  const _Float16* p = lds + r * 288 + k0 + koff;
  v16h a;
#pragma unroll
  for (int i = 0; i < 8; ++i) a[i] = p[i];
#pragma unroll
  for (int i = 0; i < 8; ++i) a[8 + i] = p[16 + i];
  return a;
}

__device__ __forceinline__ v16h load_b(const _Float16* w, int K, int n0, int k0, int lane) {
  const int n  = lane & 15;
  const int kb = (lane >> 4) << 3;              // 0 or 8
  const _Float16* p = w + (size_t)(n0 + n) * K + k0 + kb;
  v16h b;
#pragma unroll
  for (int i = 0; i < 8; ++i) b[i] = p[i];
#pragma unroll
  for (int i = 0; i < 8; ++i) b[8 + i] = p[16 + i];
  return b;
}

// Dual-env GEMM tile: one B fragment feeds two WMMAs (env0 rows 0-15, env1 16-31)
__device__ __forceinline__ void gemm16x2(const _Float16* aLds, const _Float16* w,
                                         int K, int n0, int lane, v8f& d0, v8f& d1) {
  v8f a0 = {}, a1 = {};
  for (int k0 = 0; k0 < K; k0 += 32) {
    v16h b  = load_b(w, K, n0, k0, lane);
    v16h f0 = load_a(aLds, k0, lane);
    v16h f1 = load_a(aLds + 16 * 288, k0, lane);
    a0 = WMMA16(f0, b, a0);
    a1 = WMMA16(f1, b, a1);
  }
  d0 = a0; d1 = a1;
}

__device__ __forceinline__ void store_relu_h16(_Float16* dst, int rbase, int n0,
                                               v8f d, const float* bias, int lane) {
  const int c  = (lane & 15) + n0;
  const int rb = rbase + ((lane >> 4) << 3);
  const float b = bias[c];
#pragma unroll
  for (int i = 0; i < 8; ++i) {
    float v = d[i] + b;
    dst[(rb + i) * 288 + c] = (_Float16)(v > 0.f ? v : 0.f);
  }
}

__device__ __forceinline__ void store_bias_f32(float* dst, int stride, int rbase, int n0,
                                               v8f d, const float* bias, int lane) {
  const int c  = (lane & 15) + n0;
  const int rb = rbase + ((lane >> 4) << 3);
  const float b = bias[c];
#pragma unroll
  for (int i = 0; i < 8; ++i) dst[(rb + i) * stride + c] = d[i] + b;
}

__device__ __forceinline__ float sigm(float x) { return 1.0f / (1.0f + expf(-x)); }

// ---------------------------------------------------------------------------
// Weight pack kernels (f32 -> f16, run once per call into d_ws)
// ---------------------------------------------------------------------------
__global__ void cvt_f16(_Float16* __restrict__ dst, const float* __restrict__ src, int n) {
  int i = blockIdx.x * blockDim.x + threadIdx.x;
  if (i < n) dst[i] = (_Float16)src[i];
}

__global__ void pack_qkv(_Float16* __restrict__ dst, const float* __restrict__ qw,
                         const float* __restrict__ kw, const float* __restrict__ vw,
                         const float* __restrict__ hgw) {
  int i = blockIdx.x * blockDim.x + threadIdx.x;
  if (i >= 80 * 256) return;
  int r = i >> 8, c = i & 255;
  float v = 0.f;
  if      (r < 32) v = qw[r * 256 + c];
  else if (r < 64) v = kw[(r - 32) * 256 + c];
  else if (r < 72) v = vw[(r - 64) * 256 + c];
  else if (r < 74) v = hgw[(r - 72) * 256 + c];
  dst[i] = (_Float16)v;
}

__global__ void pack_rg1(_Float16* __restrict__ dst, const float* __restrict__ w) {
  int i = blockIdx.x * blockDim.x + threadIdx.x;
  if (i >= 256 * 288) return;
  int r = i / 288, c = i - r * 288;
  dst[i] = (_Float16)(c < 264 ? w[r * 264 + c] : 0.f);
}

// ---------------------------------------------------------------------------
// Fused agent kernel: one block = two envs (32 agents), 128 threads = 4 waves
// ---------------------------------------------------------------------------
__global__ __launch_bounds__(128, 2) void fused_agent(
    const float* __restrict__ inputs, const float* __restrict__ hidden,
    const _Float16* __restrict__ wsh,
    const float* __restrict__ fc1_b, const float* __restrict__ bih, const float* __restrict__ bhh,
    const float* __restrict__ q_b, const float* __restrict__ k_b,
    const float* __restrict__ v_b, const float* __restrict__ hg_b,
    const float* __restrict__ mln_g, const float* __restrict__ mln_b,
    const float* __restrict__ fln_g, const float* __restrict__ fln_b,
    const float* __restrict__ rp_w, const float* __restrict__ rp_b,
    const float* __restrict__ rg1_b, const float* __restrict__ rg2_w, const float* __restrict__ rg2_b,
    const float* __restrict__ ph1_b, const float* __restrict__ ph2_b,
    float* __restrict__ out_logits, float* __restrict__ out_h) {
  __shared__ _Float16 sA0[ROWS * 288];   // inputs f16 -> h f16
  __shared__ _Float16 sA1[ROWS * 288];   // h_in f16 -> rg_in f16 -> fused f16
  __shared__ _Float16 sA2[ROWS * 288];   // x f16 -> rg1-hidden f16 -> ph1-hidden f16
  __shared__ float sH[ROWS * 256];       // h_in f32 -> h f32
  __shared__ float sQKV[ROWS * 80];      // q|k|v|hg outputs
  __shared__ float sScore[ROWS * 2 * 16];// (agent,head) x key
  __shared__ float sMsg[ROWS * 8];
  __shared__ float sFlat[ROWS * 8];
  __shared__ float sRG[ROWS];
  __shared__ float sB80[80];

  const int tid  = threadIdx.x;
  const int lane = tid & 31;
  const int wave = tid >> 5;
  const size_t rowbase = (size_t)blockIdx.x * ROWS;  // 32 consecutive global rows

  const _Float16* Wfc1 = wsh + OFF_FC1;
  const _Float16* Wih  = wsh + OFF_WIH;
  const _Float16* Whh  = wsh + OFF_WHH;
  const _Float16* Wqkv = wsh + OFF_QKV;
  const _Float16* Wrg1 = wsh + OFF_RG1;
  const _Float16* Wph1 = wsh + OFF_PH1;
  const _Float16* Wph2 = wsh + OFF_PH2;

  if (tid < 80) {
    float b = 0.f;
    if      (tid < 32) b = q_b[tid];
    else if (tid < 64) b = k_b[tid - 32];
    else if (tid < 72) b = v_b[tid - 64];
    else if (tid < 74) b = hg_b[tid - 72];
    sB80[tid] = b;
  }
  // ---- stage inputs / hidden (float4 = b128 loads) ----
  {
    const float4* in4 = (const float4*)inputs;
    const float4* hi4 = (const float4*)hidden;
    for (int idx = tid; idx < ROWS * 64; idx += 128) {
      int a = idx >> 6, c4 = idx & 63, c = c4 << 2;
      float4 iv = in4[(rowbase + a) * 64 + c4];
      float4 hv = hi4[(rowbase + a) * 64 + c4];
      _Float16* pa = &sA0[a * 288 + c];
      pa[0] = (_Float16)iv.x; pa[1] = (_Float16)iv.y;
      pa[2] = (_Float16)iv.z; pa[3] = (_Float16)iv.w;
      float* ph = &sH[a * 256 + c];
      ph[0] = hv.x; ph[1] = hv.y; ph[2] = hv.z; ph[3] = hv.w;
      _Float16* pb = &sA1[a * 288 + c];
      pb[0] = (_Float16)hv.x; pb[1] = (_Float16)hv.y;
      pb[2] = (_Float16)hv.z; pb[3] = (_Float16)hv.w;
    }
  }
  __syncthreads();

  // ---- fc1 + relu -> x (sA2) ----
  for (int t = wave; t < 16; t += 4) {
    v8f d0, d1;
    gemm16x2(sA0, Wfc1, 256, t * 16, lane, d0, d1);
    store_relu_h16(sA2, 0,  t * 16, d0, fc1_b, lane);
    store_relu_h16(sA2, 16, t * 16, d1, fc1_b, lane);
  }
  __syncthreads();

  // ---- GRU: 12 accumulators/tile (3 gates x ih/hh x 2 envs), B loaded once ----
  for (int t = wave; t < 16; t += 4) {
    const int c0 = t * 16;
    v8f gi[3][2] = {}, gh[3][2] = {};
    for (int k0 = 0; k0 < 256; k0 += 32) {
      v16h ax0 = load_a(sA2, k0, lane);
      v16h ax1 = load_a(sA2 + 16 * 288, k0, lane);
      v16h ah0 = load_a(sA1, k0, lane);
      v16h ah1 = load_a(sA1 + 16 * 288, k0, lane);
#pragma unroll
      for (int g = 0; g < 3; ++g) {
        v16h bI = load_b(Wih, 256, g * 256 + c0, k0, lane);
        gi[g][0] = WMMA16(ax0, bI, gi[g][0]);
        gi[g][1] = WMMA16(ax1, bI, gi[g][1]);
        v16h bH = load_b(Whh, 256, g * 256 + c0, k0, lane);
        gh[g][0] = WMMA16(ah0, bH, gh[g][0]);
        gh[g][1] = WMMA16(ah1, bH, gh[g][1]);
      }
    }
    const int c  = c0 + (lane & 15);
    const int rb = (lane >> 4) << 3;
    const float bir = bih[c], biz = bih[256 + c], bin = bih[512 + c];
    const float bhr = bhh[c], bhz = bhh[256 + c], bhn = bhh[512 + c];
#pragma unroll
    for (int v = 0; v < 2; ++v) {
#pragma unroll
      for (int i = 0; i < 8; ++i) {
        const int r_ = v * 16 + rb + i;
        float hprev = sH[r_ * 256 + c];
        float rr = sigm(gi[0][v][i] + bir + gh[0][v][i] + bhr);
        float zz = sigm(gi[1][v][i] + biz + gh[1][v][i] + bhz);
        float nn = tanhf(gi[2][v][i] + bin + rr * (gh[2][v][i] + bhn));
        float hv = (1.f - zz) * nn + zz * hprev;
        out_h[(rowbase + r_) * 256 + c] = hv;
        sH[r_ * 256 + c]  = hv;
        sA0[r_ * 288 + c] = (_Float16)hv;
      }
    }
  }
  __syncthreads();

  // ---- q|k|v|hg projection (5 tiles over 80 fused out-cols, both envs) ----
  for (int t = wave; t < 5; t += 4) {
    v8f d0, d1;
    gemm16x2(sA0, Wqkv, 256, t * 16, lane, d0, d1);
    store_bias_f32(sQKV, 80, 0,  t * 16, d0, sB80, lane);
    store_bias_f32(sQKV, 80, 16, t * 16, d1, sB80, lane);
  }
  __syncthreads();

  // ---- attention scores (per env, 2 heads, head_dim 16), diag masked ----
  for (int e = tid; e < ROWS * 32; e += 128) {
    int el  = e >> 9;                 // env 0/1
    int rem = e & 511;
    int i = rem >> 5, r2 = rem & 31, hh = r2 >> 4, j = r2 & 15;
    const int gi_ = el * 16 + i, gj = el * 16 + j;
    const float* qp = &sQKV[gi_ * 80 + hh * 16];
    const float* kp = &sQKV[gj * 80 + 32 + hh * 16];
    float s = 0.f;
#pragma unroll
    for (int d2 = 0; d2 < 16; ++d2) s += qp[d2] * kp[d2];
    s *= 0.25f;                        // 1/sqrt(16)
    if (j == i) s = NEGV;
    sScore[(gi_ * 2 + hh) * 16 + j] = s;
  }
  __syncthreads();

  // ---- top-4 mask + softmax per (agent, head): 64 rows ----
  if (tid < 64) {
    float* row = &sScore[tid * 16];
    unsigned chosen = 0;
    for (int t = 0; t < 4; ++t) {
      float best = -3e38f; int bj = 0;
      for (int j = 0; j < 16; ++j)
        if (!((chosen >> j) & 1u) && row[j] > best) { best = row[j]; bj = j; }
      chosen |= 1u << bj;
    }
    float m = -3e38f;
    for (int j = 0; j < 16; ++j) {
      if (!((chosen >> j) & 1u)) row[j] = NEGV;
      if (row[j] > m) m = row[j];
    }
    float ssum = 0.f;
    for (int j = 0; j < 16; ++j) { float e2 = expf(row[j] - m); row[j] = e2; ssum += e2; }
    float inv = 1.f / ssum;
    for (int j = 0; j < 16; ++j) row[j] *= inv;
  }
  __syncthreads();

  // ---- messages: alpha @ v  (256 entries: agent x head x vdim) ----
  for (int e = tid; e < ROWS * 8; e += 128) {
    int i = e >> 3, hh = (e >> 2) & 1, vd = e & 3;
    const int env = i >> 4;
    const float* al = &sScore[(i * 2 + hh) * 16];
    float s = 0.f;
    for (int j = 0; j < 16; ++j) s += al[j] * sQKV[(env * 16 + j) * 80 + 64 + hh * 4 + vd];
    sMsg[i * 8 + hh * 4 + vd] = s;
  }
  __syncthreads();

  // ---- per-head LN over vdim=4, then head gate (64 pairs) ----
  if (tid < 64) {
    int i = tid >> 1, hh = tid & 1;
    float* mp = &sMsg[i * 8 + hh * 4];
    float mean = 0.25f * (mp[0] + mp[1] + mp[2] + mp[3]);
    float var = 0.f;
#pragma unroll
    for (int v = 0; v < 4; ++v) { float d = mp[v] - mean; var += d * d; }
    var *= 0.25f;
    float inv = rsqrtf(var + 1e-5f);
    float gate = sigm(sQKV[i * 80 + 72 + hh]);
#pragma unroll
    for (int v = 0; v < 4; ++v)
      mp[v] = ((mp[v] - mean) * inv * mln_g[v] + mln_b[v]) * gate;
  }
  __syncthreads();

  // ---- flat LN over 8 -> sFlat (32 rows) ----
  if (tid < ROWS) {
    float mean = 0.f;
    for (int j = 0; j < 8; ++j) mean += sMsg[tid * 8 + j];
    mean *= 0.125f;
    float var = 0.f;
    for (int j = 0; j < 8; ++j) { float d = sMsg[tid * 8 + j] - mean; var += d * d; }
    var *= 0.125f;
    float inv = rsqrtf(var + 1e-5f);
    for (int j = 0; j < 8; ++j)
      sFlat[tid * 8 + j] = (sMsg[tid * 8 + j] - mean) * inv * fln_g[j] + fln_b[j];
  }
  __syncthreads();

  // ---- rg_in = [h | flat | 0pad] f16 -> sA1 (K=288) ----
  for (int idx = tid; idx < ROWS * 288; idx += 128) {
    int a = idx / 288, c = idx - a * 288;
    _Float16 v;
    if      (c < 256) v = sA0[a * 288 + c];
    else if (c < 264) v = (_Float16)sFlat[a * 8 + (c - 256)];
    else              v = (_Float16)0.f;
    sA1[idx] = v;
  }
  __syncthreads();

  // ---- rg1 (K=288) + relu -> sA2 ----
  for (int t = wave; t < 16; t += 4) {
    v8f d0, d1;
    gemm16x2(sA1, Wrg1, 288, t * 16, lane, d0, d1);
    store_relu_h16(sA2, 0,  t * 16, d0, rg1_b, lane);
    store_relu_h16(sA2, 16, t * 16, d1, rg1_b, lane);
  }
  __syncthreads();

  // ---- rg2 dot(256) + sigmoid (32 rows) ----
  if (tid < ROWS) {
    float s = rg2_b[0];
    for (int c = 0; c < 256; ++c) s += (float)sA2[tid * 288 + c] * rg2_w[c];
    sRG[tid] = sigm(s);
  }
  __syncthreads();

  // ---- residual projection (K=8) + fuse -> sA1 f16 ----
  for (int idx = tid; idx < ROWS * 256; idx += 128) {
    int a = idx >> 8, c = idx & 255;
    float r = rp_b[c];
#pragma unroll
    for (int k = 0; k < 8; ++k) r += sFlat[a * 8 + k] * rp_w[c * 8 + k];
    sA1[a * 288 + c] = (_Float16)(sH[a * 256 + c] + r * 0.5f * sRG[a]);
  }
  __syncthreads();

  // ---- ph1 + relu -> sA2 ----
  for (int t = wave; t < 16; t += 4) {
    v8f d0, d1;
    gemm16x2(sA1, Wph1, 256, t * 16, lane, d0, d1);
    store_relu_h16(sA2, 0,  t * 16, d0, ph1_b, lane);
    store_relu_h16(sA2, 16, t * 16, d1, ph1_b, lane);
  }
  __syncthreads();

  // ---- ph2 -> logits (2 tiles, both envs) ----
  for (int t = wave; t < 2; t += 4) {
    v8f d0, d1;
    gemm16x2(sA2, Wph2, 256, t * 16, lane, d0, d1);
    const int c  = (lane & 15) + t * 16;
    const int rb = (lane >> 4) << 3;
    const float b = ph2_b[c];
#pragma unroll
    for (int i = 0; i < 8; ++i) {
      out_logits[(rowbase + rb + i) * NACT + c]      = d0[i] + b;
      out_logits[(rowbase + 16 + rb + i) * NACT + c] = d1[i] + b;
    }
  }
}

// ---------------------------------------------------------------------------
extern "C" void kernel_launch(void* const* d_in, const int* in_sizes, int n_in,
                              void* d_out, int out_size, void* d_ws, size_t ws_size,
                              hipStream_t stream) {
  // params in setup_inputs() dict order
  const float* fc1_w   = (const float*)d_in[0];
  const float* fc1_b   = (const float*)d_in[1];
  const float* gru_wih = (const float*)d_in[2];
  const float* gru_bih = (const float*)d_in[3];
  const float* gru_whh = (const float*)d_in[4];
  const float* gru_bhh = (const float*)d_in[5];
  const float* q_w  = (const float*)d_in[6];
  const float* q_b  = (const float*)d_in[7];
  const float* k_w  = (const float*)d_in[8];
  const float* k_b  = (const float*)d_in[9];
  const float* v_w  = (const float*)d_in[10];
  const float* v_b  = (const float*)d_in[11];
  const float* hg_w = (const float*)d_in[12];
  const float* hg_b = (const float*)d_in[13];
  const float* mln_g = (const float*)d_in[14];
  const float* mln_b = (const float*)d_in[15];
  const float* fln_g = (const float*)d_in[16];
  const float* fln_b = (const float*)d_in[17];
  const float* rp_w  = (const float*)d_in[18];
  const float* rp_b  = (const float*)d_in[19];
  const float* rg1_w = (const float*)d_in[20];
  const float* rg1_b = (const float*)d_in[21];
  const float* rg2_w = (const float*)d_in[22];
  const float* rg2_b = (const float*)d_in[23];
  const float* ph1_w = (const float*)d_in[24];
  const float* ph1_b = (const float*)d_in[25];
  const float* ph2_w = (const float*)d_in[26];
  const float* ph2_b = (const float*)d_in[27];
  const float* inputs = (const float*)d_in[28];
  const float* hidden = (const float*)d_in[29];

  _Float16* ws = (_Float16*)d_ws;   // ~1.25 MB of f16 weights
  float* out_logits = (float*)d_out;
  float* out_h = out_logits + (size_t)BS * NAG * NACT;

  auto blocks = [](int n) { return (n + 255) / 256; };
  cvt_f16<<<blocks(256 * 256), 256, 0, stream>>>(ws + OFF_FC1, fc1_w, 256 * 256);
  cvt_f16<<<blocks(768 * 256), 256, 0, stream>>>(ws + OFF_WIH, gru_wih, 768 * 256);
  cvt_f16<<<blocks(768 * 256), 256, 0, stream>>>(ws + OFF_WHH, gru_whh, 768 * 256);
  pack_qkv<<<blocks(80 * 256), 256, 0, stream>>>(ws + OFF_QKV, q_w, k_w, v_w, hg_w);
  pack_rg1<<<blocks(256 * 288), 256, 0, stream>>>(ws + OFF_RG1, rg1_w);
  cvt_f16<<<blocks(256 * 256), 256, 0, stream>>>(ws + OFF_PH1, ph1_w, 256 * 256);
  cvt_f16<<<blocks(32 * 256), 256, 0, stream>>>(ws + OFF_PH2, ph2_w, 32 * 256);

  fused_agent<<<BS / EPB, 128, 0, stream>>>(
      inputs, hidden, ws,
      fc1_b, gru_bih, gru_bhh,
      q_b, k_b, v_b, hg_b,
      mln_g, mln_b, fln_g, fln_b,
      rp_w, rp_b, rg1_b, rg2_w, rg2_b,
      ph1_b, ph2_b,
      out_logits, out_h);
}
